// TinyRNN_86543591014816
// MI455X (gfx1250) — matrix operand order repaired and checked by reading, not compiled, not run on hardware
//
#include <hip/hip_runtime.h>

#define T_STEPS 4096
#define IN_DIM  2048
#define HID_DIM 2048
#define OUT_DIM 2048
#define NBLK_SCAN 64

typedef __attribute__((ext_vector_type(16))) __bf16 v16bf;
typedef __attribute__((ext_vector_type(8)))  float  v8f;
typedef __attribute__((ext_vector_type(4)))  unsigned int u32x4;
typedef __attribute__((ext_vector_type(4)))  float  f32x4;

union FragAB { v16bf v; u32x4 q[2]; };

__device__ __forceinline__ unsigned short f2bf(float f) {
    unsigned int u = __float_as_uint(f);
    unsigned int r = (u + 0x7FFFu + ((u >> 16) & 1u)) >> 16;   // RNE
    return (unsigned short)r;
}

// ---------------------------------------------------------------- conversion
__global__ void f32_to_bf16_kernel(const float* __restrict__ in,
                                   unsigned short* __restrict__ out, size_t n) {
    size_t i = (size_t)blockIdx.x * blockDim.x + threadIdx.x;
    size_t stride = (size_t)gridDim.x * blockDim.x;
    for (; i < n; i += stride) out[i] = f2bf(in[i]);
}

// ---------------------------------------------------------------- init
__global__ void init_kernel(const float* __restrict__ h_prev,
                            float* __restrict__ hbuf, unsigned int* __restrict__ bar) {
    int i = blockIdx.x * blockDim.x + threadIdx.x;
    if (i < HID_DIM) { hbuf[i] = h_prev[i]; hbuf[HID_DIM + i] = 0.0f; }
    if (i < 4) bar[i] = 0u;
}

// ---------------------------------------------------------------- WMMA GEMM
// C[m][n] = sum_k A[m][k]*B[n][k] + bias[n]    (A:[M,K] bf16, B:[N,K] bf16)
// wave tile 64x64 (4x4 WMMA frags), block = 8 waves (2x4) -> 128x256 tile.
__global__ __launch_bounds__(256) void gemm_bf16_nt(
    const unsigned short* __restrict__ A,
    const unsigned short* __restrict__ B,
    const float* __restrict__ bias,
    float* __restrict__ C,
    int M, int N, int K) {
    const int lane = threadIdx.x & 31;
    const int wave = threadIdx.x >> 5;
    const int wm = wave >> 2;            // 0..1
    const int wn = wave & 3;             // 0..3
    const int m0 = blockIdx.x * 128 + wm * 64;
    const int n0 = blockIdx.y * 256 + wn * 64;
    const int lm = lane & 15;
    const int hi = lane >> 4;            // 0 | 1
    const int ka_off = hi * 8;           // A: lanes>=16 carry K+8 / K+24
    const int kb_off = hi * 8;           // B: lanes>=16 carry K+8 / K+24

    v8f acc[4][4];
    #pragma unroll
    for (int j = 0; j < 4; ++j) {
        float bv = bias[n0 + j * 16 + lm];
        #pragma unroll
        for (int i = 0; i < 4; ++i)
            #pragma unroll
            for (int r = 0; r < 8; ++r) acc[i][j][r] = bv;
    }

    const unsigned short* aBase[4];
    const unsigned short* bBase[4];
    #pragma unroll
    for (int i = 0; i < 4; ++i) aBase[i] = A + (size_t)(m0 + i * 16 + lm) * K;
    #pragma unroll
    for (int j = 0; j < 4; ++j) bBase[j] = B + (size_t)(n0 + j * 16 + lm) * K;

    for (int k0 = 0; k0 < K; k0 += 32) {
        FragAB af[4], bfr[4];
        #pragma unroll
        for (int i = 0; i < 4; ++i) {
            const unsigned short* p = aBase[i] + k0 + ka_off;
            af[i].q[0] = *(const u32x4*)(p);        // K 0..7  (or 8..15)
            af[i].q[1] = *(const u32x4*)(p + 16);   // K 16..23 (or 24..31)
        }
        #pragma unroll
        for (int j = 0; j < 4; ++j) {
            const unsigned short* p = bBase[j] + k0 + kb_off;
            bfr[j].q[0] = *(const u32x4*)(p);       // K 0..7  (or 8..15)
            bfr[j].q[1] = *(const u32x4*)(p + 16);  // K 16..23 (or 24..31)
        }
        #pragma unroll
        for (int i = 0; i < 4; ++i)
            #pragma unroll
            for (int j = 0; j < 4; ++j)
                acc[i][j] = __builtin_amdgcn_wmma_f32_16x16x32_bf16(
                    false, af[i].v, false, bfr[j].v, (short)0, acc[i][j],
                    false, false);
    }

    #pragma unroll
    for (int i = 0; i < 4; ++i) {
        int mrow = m0 + i * 16 + hi * 8;
        #pragma unroll
        for (int j = 0; j < 4; ++j) {
            int n = n0 + j * 16 + lm;
            #pragma unroll
            for (int r = 0; r < 8; ++r)
                C[(size_t)(mrow + r) * N + n] = acc[i][j][r];
        }
    }
}

// ---------------------------------------------------------------- grid barrier
__device__ __forceinline__ void grid_sync(unsigned int* counter, unsigned int* gen,
                                          unsigned int nblk) {
    __syncthreads();
    if (threadIdx.x == 0) {
        __threadfence();
        unsigned int g = __hip_atomic_load(gen, __ATOMIC_RELAXED,
                                           __HIP_MEMORY_SCOPE_AGENT);
        unsigned int arrived = __hip_atomic_fetch_add(counter, 1u, __ATOMIC_ACQ_REL,
                                                      __HIP_MEMORY_SCOPE_AGENT) + 1u;
        if (arrived == nblk) {
            __hip_atomic_store(counter, 0u, __ATOMIC_RELAXED,
                               __HIP_MEMORY_SCOPE_AGENT);
            __hip_atomic_fetch_add(gen, 1u, __ATOMIC_RELEASE,
                                   __HIP_MEMORY_SCOPE_AGENT);
        } else {
            while (__hip_atomic_load(gen, __ATOMIC_ACQUIRE,
                                     __HIP_MEMORY_SCOPE_AGENT) == g)
                __builtin_amdgcn_s_sleep(2);
        }
    }
    __syncthreads();
}

// ---------------------------------------------------------------- recurrence
// Persistent kernel: 64 blocks x 256 threads, 32 outputs/block, 8 lanes/output.
// Each lane pins its 256 bf16 W_hh weights in 128 VGPRs (64blk*256ln*512B = 8MB
// = whole matrix register-resident) -> per step only the 8KB h vector moves.
__global__ __launch_bounds__(256, 1) void rnn_scan_kernel(
    const float* __restrict__ xproj,                 // [T, HID]
    const unsigned short* __restrict__ Whhbf,        // [HID, HID] bf16
    const float* __restrict__ bhh,                   // [HID]
    float* __restrict__ hbuf,                        // [2, HID]
    unsigned short* __restrict__ Hbf,                // [T, HID] bf16
    float* __restrict__ h_final,                     // [HID]
    unsigned int* __restrict__ bar) {
    const int tid = threadIdx.x;
    const int j   = blockIdx.x * 32 + (tid >> 3);    // output row of W_hh
    const int sub = tid & 7;                         // k-slice 256 wide
    const float bj = bhh[j];

    // Pin this lane's weight slice in registers: 32 x u32x4 = 128 VGPRs.
    u32x4 wreg[32];
    {
        const u32x4* wp =
            (const u32x4*)(Whhbf + (size_t)j * HID_DIM + sub * 256);
        #pragma unroll
        for (int q = 0; q < 32; ++q) wreg[q] = wp[q];
    }

    for (int t = 0; t < T_STEPS; ++t) {
        const f32x4* hc =
            (const f32x4*)(hbuf + (size_t)(t & 1) * HID_DIM + sub * 256);
        float* hnext = hbuf + (size_t)((t + 1) & 1) * HID_DIM;
        float sum = 0.0f;
        #pragma unroll
        for (int q = 0; q < 32; ++q) {
            f32x4 h0 = hc[2 * q];
            f32x4 h1 = hc[2 * q + 1];
            u32x4 w  = wreg[q];
            // packed bf16 pair per u32: lo16 = k, hi16 = k+1; bf16->f32 = <<16
            sum = fmaf(__uint_as_float(w[0] << 16),          h0[0], sum);
            sum = fmaf(__uint_as_float(w[0] & 0xffff0000u),  h0[1], sum);
            sum = fmaf(__uint_as_float(w[1] << 16),          h0[2], sum);
            sum = fmaf(__uint_as_float(w[1] & 0xffff0000u),  h0[3], sum);
            sum = fmaf(__uint_as_float(w[2] << 16),          h1[0], sum);
            sum = fmaf(__uint_as_float(w[2] & 0xffff0000u),  h1[1], sum);
            sum = fmaf(__uint_as_float(w[3] << 16),          h1[2], sum);
            sum = fmaf(__uint_as_float(w[3] & 0xffff0000u),  h1[3], sum);
        }
        sum += __shfl_xor(sum, 1);
        sum += __shfl_xor(sum, 2);
        sum += __shfl_xor(sum, 4);
        if (sub == 0) {
            float y = tanhf(sum + bj + xproj[(size_t)t * HID_DIM + j]);
            hnext[j] = y;
            Hbf[(size_t)t * HID_DIM + j] = f2bf(y);
            if (t == T_STEPS - 1) h_final[j] = y;
        }
        grid_sync(bar, bar + 1, (unsigned int)gridDim.x);
    }
}

// ---------------------------------------------------------------- launch
extern "C" void kernel_launch(void* const* d_in, const int* in_sizes, int n_in,
                              void* d_out, int out_size, void* d_ws, size_t ws_size,
                              hipStream_t stream) {
    const float* x_seq  = (const float*)d_in[0];
    const float* h_prev = (const float*)d_in[1];
    const float* W_ih   = (const float*)d_in[2];
    const float* b_ih   = (const float*)d_in[3];
    const float* W_hh   = (const float*)d_in[4];
    const float* b_hh   = (const float*)d_in[5];
    const float* W_fc   = (const float*)d_in[6];
    const float* b_fc   = (const float*)d_in[7];
    float* out = (float*)d_out;    // [T*OUT] logits, then [HID] h_final

    char* ws = (char*)d_ws;
    size_t off = 0;
    auto carve = [&](size_t bytes) {
        char* p = ws + off;
        off = (off + bytes + 255) & ~(size_t)255;
        return p;
    };
    unsigned short* xbf   = (unsigned short*)carve((size_t)T_STEPS * IN_DIM * 2);
    unsigned short* wihbf = (unsigned short*)carve((size_t)HID_DIM * IN_DIM * 2);
    unsigned short* wfcbf = (unsigned short*)carve((size_t)OUT_DIM * HID_DIM * 2);
    unsigned short* whhbf = (unsigned short*)carve((size_t)HID_DIM * HID_DIM * 2);
    float*          xproj = (float*)carve((size_t)T_STEPS * HID_DIM * 4);
    unsigned short* Hbf   = (unsigned short*)carve((size_t)T_STEPS * HID_DIM * 2);
    float*          hbuf  = (float*)carve((size_t)2 * HID_DIM * 4);
    unsigned int*   bar   = (unsigned int*)carve(256);

    // 1) bf16 copies of GEMM/recurrence operands
    f32_to_bf16_kernel<<<4096, 256, 0, stream>>>(x_seq, xbf,
                                                 (size_t)T_STEPS * IN_DIM);
    f32_to_bf16_kernel<<<2048, 256, 0, stream>>>(W_ih, wihbf,
                                                 (size_t)HID_DIM * IN_DIM);
    f32_to_bf16_kernel<<<2048, 256, 0, stream>>>(W_fc, wfcbf,
                                                 (size_t)OUT_DIM * HID_DIM);
    f32_to_bf16_kernel<<<2048, 256, 0, stream>>>(W_hh, whhbf,
                                                 (size_t)HID_DIM * HID_DIM);
    // 2) h0 + barrier init
    init_kernel<<<8, 256, 0, stream>>>(h_prev, hbuf, bar);

    // 3) x_proj = x_seq @ W_ih^T + b_ih   (WMMA bf16 -> fp32)
    gemm_bf16_nt<<<dim3(T_STEPS / 128, HID_DIM / 256), 256, 0, stream>>>(
        xbf, wihbf, b_ih, xproj, T_STEPS, HID_DIM, IN_DIM);

    // 4) sequential recurrence (persistent, register-resident W_hh,
    //    grid-barrier per step)
    rnn_scan_kernel<<<NBLK_SCAN, 256, 0, stream>>>(
        xproj, whhbf, b_hh, hbuf, Hbf, out + (size_t)T_STEPS * OUT_DIM, bar);

    // 5) logits = H @ W_fc^T + b_fc   (batched over all timesteps, WMMA)
    gemm_bf16_nt<<<dim3(T_STEPS / 128, OUT_DIM / 256), 256, 0, stream>>>(
        Hbf, wfcbf, b_fc, out, T_STEPS, OUT_DIM, HID_DIM);
}
